// ChromaSelfAttention_63737314673112
// MI455X (gfx1250) — compile-verified, matrix operand order checked by reading, not run on hardware
//
#include <hip/hip_runtime.h>

typedef _Float16 h16;
typedef __attribute__((ext_vector_type(16))) _Float16 v16h;
typedef __attribute__((ext_vector_type(8)))  _Float16 v8h;
typedef __attribute__((ext_vector_type(8)))  float    v8f;

#define WMMA16(a,b,c) __builtin_amdgcn_wmma_f32_16x16x32_f16(false,(a),false,(b),(short)0,(c),false,false)

#define L_SEQ  2048
#define DMODEL 3072
#define NHEADS 24
#define DHEAD  128
#define KT     32
#define LDP    40    // LDS K-tile row stride (elements): 80B rows, 16B aligned, conflict-spread
#define LDA    136   // LDS 128-wide row stride: 272B rows, 16B aligned

#define AS1 __attribute__((address_space(1)))
#define AS3 __attribute__((address_space(3)))

// b128 transfer unit type exactly as the async-LDS builtins expect (from the clang diagnostic):
typedef int b128i __attribute__((vector_size(16)));

#if __has_builtin(__builtin_amdgcn_global_load_async_to_lds_b128) && \
    __has_builtin(__builtin_amdgcn_s_wait_asynccnt)
#define USE_ASYNC_LDS 1
#else
#define USE_ASYNC_LDS 0
#endif

// Load one 16x32 f16 A/B fragment (ISA 7.12.2 layout) from row-major [16][ld]:
// lanes 0-15 = rows, K packed {0..7,16..23}; lanes 16-31 same rows, K {8..15,24..31}.
// Two b128 loads per lane.
__device__ __forceinline__ v16h load_frag(const h16* __restrict__ base, int ld) {
  const int lane = threadIdx.x & 31;
  const h16* p = base + (lane & 15) * ld + 8 * (lane >> 4);
  v8h lo = *(const v8h*)(p);
  v8h hi = *(const v8h*)(p + 16);
  return __builtin_shufflevector(lo, hi, 0,1,2,3,4,5,6,7,8,9,10,11,12,13,14,15);
}

// ---------------- QKV projection: Y = X@W + b, optional fused per-head RMS norm, f16 head-major out
__global__ __launch_bounds__(256, 1)
void proj_kernel(const float* __restrict__ X, const float* __restrict__ W,
                 const float* __restrict__ bias, const float* __restrict__ normw,
                 float extra_scale, h16* __restrict__ out,
                 int M, int K, int N) {
  __shared__ h16 Xs[128 * LDP];
  __shared__ h16 Wt[128 * LDP];
  __shared__ float rowstat[128];

  const int t    = threadIdx.x;
  const int wave = t >> 5;
  const int wm   = wave >> 1;        // 0..3 -> 32-row strip
  const int wn   = wave & 1;         // 0..1 -> 64-col strip
  const int ln   = t & 15;
  const int half = (t >> 4) & 1;
  const int m0   = blockIdx.x * 128;
  const int n0   = blockIdx.y * 128;

  const v8f vzero = {0.f,0.f,0.f,0.f,0.f,0.f,0.f,0.f};
  v8f acc[2][4];
#pragma unroll
  for (int i = 0; i < 2; ++i)
#pragma unroll
    for (int j = 0; j < 4; ++j) acc[i][j] = vzero;

  float xr[16], wr[16];
  auto load_regs = [&](int k0) {
#pragma unroll
    for (int i = 0; i < 16; ++i) {
      int idx = t + 256 * i;                 // 0..4095
      int r = idx >> 5, c = idx & 31;
      xr[i] = X[(size_t)(m0 + r) * K + k0 + c];
    }
#pragma unroll
    for (int i = 0; i < 16; ++i) {
      int idx = t + 256 * i;
      int kk = idx >> 7, n = idx & 127;
      wr[i] = W[(size_t)(k0 + kk) * N + n0 + n];
    }
  };

  load_regs(0);                               // prologue: tile 0 in regs
  for (int k0 = 0; k0 < K; k0 += KT) {
    __syncthreads();                          // previous compute done reading LDS
#pragma unroll
    for (int i = 0; i < 16; ++i) {            // regs -> LDS (f32 -> f16)
      int idx = t + 256 * i;
      Xs[(idx >> 5) * LDP + (idx & 31)]  = (h16)xr[i];
      Wt[(idx & 127) * LDP + (idx >> 7)] = (h16)wr[i];
    }
    __syncthreads();

    if (k0 + KT < K) load_regs(k0 + KT);      // overlap next-tile loads with WMMA
    if (k0 + 2 * KT < K) {                    // speculative prefetch of tile after next
      __builtin_prefetch(X + (size_t)(m0 + (t & 127)) * K + k0 + 2 * KT, 0, 1);
      __builtin_prefetch(W + (size_t)(k0 + 2 * KT + (t & 31)) * N + n0, 0, 1);
    }

    v16h af[2], bf[4];
#pragma unroll
    for (int i = 0; i < 2; ++i) af[i] = load_frag(&Xs[(wm * 32 + i * 16) * LDP], LDP);
#pragma unroll
    for (int j = 0; j < 4; ++j) bf[j] = load_frag(&Wt[(wn * 64 + j * 16) * LDP], LDP);
#pragma unroll
    for (int i = 0; i < 2; ++i)
#pragma unroll
      for (int j = 0; j < 4; ++j)
        acc[i][j] = WMMA16(af[i], bf[j], acc[i][j]);
  }

  // bias
#pragma unroll
  for (int j = 0; j < 4; ++j) {
    float b = bias[n0 + wn * 64 + j * 16 + ln];
#pragma unroll
    for (int i = 0; i < 2; ++i)
#pragma unroll
      for (int r = 0; r < 8; ++r) acc[i][j][r] += b;
  }

  if (normw) {  // fused per-head RMS norm (tile cols == one full head dim)
    if (t < 128) rowstat[t] = 0.f;
    __syncthreads();
#pragma unroll
    for (int i = 0; i < 2; ++i)
#pragma unroll
      for (int r = 0; r < 8; ++r) {
        float s = 0.f;
#pragma unroll
        for (int j = 0; j < 4; ++j) s += acc[i][j][r] * acc[i][j][r];
#pragma unroll
        for (int off = 1; off < 16; off <<= 1) s += __shfl_xor(s, off, 32);
        if (ln == 0) atomicAdd(&rowstat[wm * 32 + i * 16 + r + 8 * half], s);
      }
    __syncthreads();
    float rs[2][8];
#pragma unroll
    for (int i = 0; i < 2; ++i)
#pragma unroll
      for (int r = 0; r < 8; ++r)
        rs[i][r] = rsqrtf(rowstat[wm * 32 + i * 16 + r + 8 * half] * (1.f / 128.f) + 1e-6f) * extra_scale;
#pragma unroll
    for (int j = 0; j < 4; ++j) {
      float w = normw[wn * 64 + j * 16 + ln];
#pragma unroll
      for (int i = 0; i < 2; ++i)
#pragma unroll
        for (int r = 0; r < 8; ++r) acc[i][j][r] *= rs[i][r] * w;
    }
  }

  // store head-major [head][M][128], head == blockIdx.y
  h16* oh = out + (size_t)blockIdx.y * M * DHEAD;
#pragma unroll
  for (int i = 0; i < 2; ++i)
#pragma unroll
    for (int r = 0; r < 8; ++r) {
      int row = m0 + wm * 32 + i * 16 + r + 8 * half;
#pragma unroll
      for (int j = 0; j < 4; ++j)
        oh[(size_t)row * DHEAD + wn * 64 + j * 16 + ln] = (h16)acc[i][j][r];
    }
}

// ---------------- Flash attention: one (head, 128-row q block) per workgroup, 16 q rows per wave
#if USE_ASYNC_LDS
// Issue 16 async b128 DMA instructions per wave: K rows -> ksBuf (padded), V rows -> vsBuf (dense).
__device__ __forceinline__ void stage_kv_async(const h16* Kblk, const h16* Vblk,
                                               h16* ksBuf, h16* vsBuf, int t) {
#pragma unroll
  for (int i = 0; i < 8; ++i) {               // 256 thr * 8 elem * 8 = 16384 elems
    int e = t * 8 + 2048 * i;
    int j = e >> 7, d = e & 127;
    __builtin_amdgcn_global_load_async_to_lds_b128(
        (AS1 b128i*)(Kblk + j * DHEAD + d), (AS3 b128i*)(ksBuf + j * LDA + d), 0, 0);
    __builtin_amdgcn_global_load_async_to_lds_b128(
        (AS1 b128i*)(Vblk + j * DHEAD + d), (AS3 b128i*)(vsBuf + j * DHEAD + d), 0, 0);
  }
}
#endif

__global__ __launch_bounds__(256, 1)
void attn_kernel(const h16* __restrict__ Q, const h16* __restrict__ Kb,
                 const h16* __restrict__ Vb, h16* __restrict__ O) {
#if USE_ASYNC_LDS
  __shared__ h16 Ks[2 * 128 * LDA];    // double-buffered K rows (== K^T B-fragment layout)
  __shared__ h16 Vs[2 * 128 * DHEAD];  // double-buffered V rows (DMA landing pad)
#else
  __shared__ h16 Ks[128 * LDA];
#endif
  __shared__ h16 Vt[128 * LDA];        // V transposed [d][j]
  __shared__ h16 Ps[8 * 16 * LDA];     // per-wave P strips

  const int h    = blockIdx.y;
  const int t    = threadIdx.x;
  const int wave = t >> 5;
  const int ln   = t & 15;
  const int half = (t >> 4) & 1;

  const h16* Qh = Q + (size_t)h * L_SEQ * DHEAD;
  const h16* Kh = Kb + (size_t)h * L_SEQ * DHEAD;
  const h16* Vh = Vb + (size_t)h * L_SEQ * DHEAD;

  const int qrow0 = blockIdx.x * 128 + wave * 16;

  v16h qf[4];
#pragma unroll
  for (int kt = 0; kt < 4; ++kt)
    qf[kt] = load_frag(Qh + (size_t)qrow0 * DHEAD + kt * 32, DHEAD);

  const v8f vzero = {0.f,0.f,0.f,0.f,0.f,0.f,0.f,0.f};
  v8f oacc[8];
#pragma unroll
  for (int d = 0; d < 8; ++d) oacc[d] = vzero;
  float m_run[8], l_run[8];
#pragma unroll
  for (int r = 0; r < 8; ++r) { m_run[r] = -1e30f; l_run[r] = 0.f; }

  const int NBLK = L_SEQ / 128;
#if USE_ASYNC_LDS
  stage_kv_async(Kh, Vh, Ks, Vs, t);          // prologue: block 0 -> buffer 0
#endif

  for (int jb = 0; jb < NBLK; ++jb) {
#if USE_ASYNC_LDS
    const int cur = jb & 1;
    if (jb + 1 < NBLK) {
      stage_kv_async(Kh + (size_t)(jb + 1) * 128 * DHEAD,
                     Vh + (size_t)(jb + 1) * 128 * DHEAD,
                     Ks + (cur ^ 1) * 128 * LDA, Vs + (cur ^ 1) * 128 * DHEAD, t);
      __builtin_amdgcn_s_wait_asynccnt(16);   // drain only the 16 older (current-block) DMAs
    } else {
      __builtin_amdgcn_s_wait_asynccnt(0);
    }
    __syncthreads();
    const h16* KsCur = Ks + cur * 128 * LDA;
    const h16* VsCur = Vs + cur * 128 * DHEAD;
    // LDS->LDS transpose Vs[cur] -> Vt (consumed after the P barrier below)
#pragma unroll
    for (int i = 0; i < 16; ++i) {
      int e = t * 4 + 1024 * i;
      int j = e >> 7, d = e & 127;
      uint2 vv = *(const uint2*)&VsCur[j * DHEAD + d];
      const h16* pv = (const h16*)&vv;
#pragma unroll
      for (int q = 0; q < 4; ++q) Vt[(d + q) * LDA + j] = pv[q];
    }
#else
    __syncthreads();
    const h16* Kblk = Kh + (size_t)jb * 128 * DHEAD;
    const h16* Vblk = Vh + (size_t)jb * 128 * DHEAD;
#pragma unroll
    for (int i = 0; i < 16; ++i) {            // 256 thr x 4 elem x 16 = 16384
      int e = t * 4 + 1024 * i;
      int j = e >> 7, d = e & 127;
      *(uint2*)&Ks[j * LDA + d] = *(const uint2*)&Kblk[j * DHEAD + d];
      uint2 vv = *(const uint2*)&Vblk[j * DHEAD + d];
      const h16* pv = (const h16*)&vv;
#pragma unroll
      for (int q = 0; q < 4; ++q) Vt[(d + q) * LDA + j] = pv[q];
    }
    __syncthreads();
    const h16* KsCur = Ks;
#endif

    // S = Q K^T  (f32 accum)
    v8f sacc[8];
#pragma unroll
    for (int jt = 0; jt < 8; ++jt) {
      sacc[jt] = vzero;
#pragma unroll
      for (int kt = 0; kt < 4; ++kt) {
        v16h bf = load_frag(&KsCur[(jt * 16) * LDA + kt * 32], LDA);
        sacc[jt] = WMMA16(qf[kt], bf, sacc[jt]);
      }
    }

    // online softmax, row-wise over 128 cols (8 tiles x 16 lanes per half)
#pragma unroll
    for (int r = 0; r < 8; ++r) {
      float mx = sacc[0][r];
#pragma unroll
      for (int jt = 1; jt < 8; ++jt) mx = fmaxf(mx, sacc[jt][r]);
#pragma unroll
      for (int off = 1; off < 16; off <<= 1) mx = fmaxf(mx, __shfl_xor(mx, off, 32));
      float mnew = fmaxf(m_run[r], mx);
      float corr = __expf(m_run[r] - mnew);
      float rsum = 0.f;
#pragma unroll
      for (int jt = 0; jt < 8; ++jt) {
        float p = __expf(sacc[jt][r] - mnew);
        sacc[jt][r] = p;
        rsum += p;
      }
#pragma unroll
      for (int off = 1; off < 16; off <<= 1) rsum += __shfl_xor(rsum, off, 32);
      l_run[r] = l_run[r] * corr + rsum;
      m_run[r] = mnew;
#pragma unroll
      for (int d = 0; d < 8; ++d) oacc[d][r] *= corr;
    }

    // P -> LDS strip (C-layout -> A-layout relayout)
    h16* Pw = &Ps[wave * 16 * LDA];
#pragma unroll
    for (int r = 0; r < 8; ++r)
#pragma unroll
      for (int jt = 0; jt < 8; ++jt)
        Pw[(r + 8 * half) * LDA + jt * 16 + ln] = (h16)sacc[jt][r];
    __syncthreads();

    // O += P V
    v16h pf[4];
#pragma unroll
    for (int j2 = 0; j2 < 4; ++j2) pf[j2] = load_frag(Pw + j2 * 32, LDA);
#pragma unroll
    for (int dt = 0; dt < 8; ++dt)
#pragma unroll
      for (int j2 = 0; j2 < 4; ++j2) {
        v16h vf = load_frag(&Vt[(dt * 16) * LDA + j2 * 32], LDA);
        oacc[dt] = WMMA16(pf[j2], vf, oacc[dt]);
      }
#if USE_ASYNC_LDS
    __syncthreads();   // all reads of cur buffers done before next iteration re-issues DMA
#endif
  }

  // epilogue: divide by l, store [L][H*Dh] row-major f16
#pragma unroll
  for (int r = 0; r < 8; ++r) {
    float inv = 1.f / l_run[r];
    int grow = qrow0 + r + 8 * half;
#pragma unroll
    for (int dt = 0; dt < 8; ++dt)
      O[(size_t)grow * DMODEL + h * DHEAD + dt * 16 + ln] = (h16)(oacc[dt][r] * inv);
  }
}

// ---------------- Output projection: f16 A x f32 W -> f32 out + bias
__global__ __launch_bounds__(256, 1)
void oproj_kernel(const h16* __restrict__ A, const float* __restrict__ W,
                  const float* __restrict__ bias, float* __restrict__ out,
                  int M, int K, int N) {
  __shared__ h16 Xs[128 * LDP];
  __shared__ h16 Wt[128 * LDP];

  const int t    = threadIdx.x;
  const int wave = t >> 5;
  const int wm   = wave >> 1;
  const int wn   = wave & 1;
  const int ln   = t & 15;
  const int half = (t >> 4) & 1;
  const int m0   = blockIdx.x * 128;
  const int n0   = blockIdx.y * 128;

  const v8f vzero = {0.f,0.f,0.f,0.f,0.f,0.f,0.f,0.f};
  v8f acc[2][4];
#pragma unroll
  for (int i = 0; i < 2; ++i)
#pragma unroll
    for (int j = 0; j < 4; ++j) acc[i][j] = vzero;

  unsigned ar[8];
  float wr[16];
  auto load_regs = [&](int k0) {
#pragma unroll
    for (int i = 0; i < 8; ++i) {             // A is f16: 32-bit pair loads
      int idx = t + 256 * i;                  // 0..2047
      int r = idx >> 4, c = (idx & 15) * 2;
      ar[i] = *(const unsigned*)&A[(size_t)(m0 + r) * K + k0 + c];
    }
#pragma unroll
    for (int i = 0; i < 16; ++i) {
      int idx = t + 256 * i;
      int kk = idx >> 7, n = idx & 127;
      wr[i] = W[(size_t)(k0 + kk) * N + n0 + n];
    }
  };

  load_regs(0);
  for (int k0 = 0; k0 < K; k0 += KT) {
    __syncthreads();
#pragma unroll
    for (int i = 0; i < 8; ++i) {
      int idx = t + 256 * i;
      *(unsigned*)&Xs[(idx >> 4) * LDP + (idx & 15) * 2] = ar[i];
    }
#pragma unroll
    for (int i = 0; i < 16; ++i) {
      int idx = t + 256 * i;
      Wt[(idx & 127) * LDP + (idx >> 7)] = (h16)wr[i];
    }
    __syncthreads();

    if (k0 + KT < K) load_regs(k0 + KT);
    if (k0 + 2 * KT < K) {
      __builtin_prefetch(A + (size_t)(m0 + (t & 127)) * K + k0 + 2 * KT, 0, 1);
      __builtin_prefetch(W + (size_t)(k0 + 2 * KT + (t & 31)) * N + n0, 0, 1);
    }

    v16h af[2], bf[4];
#pragma unroll
    for (int i = 0; i < 2; ++i) af[i] = load_frag(&Xs[(wm * 32 + i * 16) * LDP], LDP);
#pragma unroll
    for (int j = 0; j < 4; ++j) bf[j] = load_frag(&Wt[(wn * 64 + j * 16) * LDP], LDP);
#pragma unroll
    for (int i = 0; i < 2; ++i)
#pragma unroll
      for (int j = 0; j < 4; ++j)
        acc[i][j] = WMMA16(af[i], bf[j], acc[i][j]);
  }

#pragma unroll
  for (int j = 0; j < 4; ++j) {
    int col = n0 + wn * 64 + j * 16 + ln;
    float b = bias[col];
#pragma unroll
    for (int i = 0; i < 2; ++i)
#pragma unroll
      for (int r = 0; r < 8; ++r) {
        int row = m0 + wm * 32 + i * 16 + r + 8 * half;
        out[(size_t)row * N + col] = acc[i][j][r] + b;
      }
  }
}

extern "C" void kernel_launch(void* const* d_in, const int* in_sizes, int n_in,
                              void* d_out, int out_size, void* d_ws, size_t ws_size,
                              hipStream_t stream) {
  const float* x_q = (const float*)d_in[0];
  const float* x_k = (const float*)d_in[1];
  const float* x_v = (const float*)d_in[2];
  const float* Wq  = (const float*)d_in[3];
  const float* bq  = (const float*)d_in[4];
  const float* Wk  = (const float*)d_in[5];
  const float* bk  = (const float*)d_in[6];
  const float* Wv  = (const float*)d_in[7];
  const float* bv  = (const float*)d_in[8];
  const float* Wo  = (const float*)d_in[9];
  const float* bo  = (const float*)d_in[10];
  const float* nqw = (const float*)d_in[11];
  const float* nkw = (const float*)d_in[12];

  const size_t per = (size_t)NHEADS * L_SEQ * DHEAD;   // f16 elements per tensor
  if (ws_size < 4 * per * sizeof(h16)) return;
  h16* qb = (h16*)d_ws;
  h16* kb = qb + per;
  h16* vb = kb + per;
  h16* ab = vb + per;

  dim3 blk(256);
  dim3 gproj(L_SEQ / 128, DMODEL / 128);               // 16 x 24
  const float qscale = 0.08838834764831845f;           // 1/sqrt(128), folded into Q

  proj_kernel<<<gproj, blk, 0, stream>>>(x_q, Wq, bq, nqw, qscale, qb, L_SEQ, DMODEL, DMODEL);
  proj_kernel<<<gproj, blk, 0, stream>>>(x_k, Wk, bk, nkw, 1.0f,  kb, L_SEQ, DMODEL, DMODEL);
  proj_kernel<<<gproj, blk, 0, stream>>>(x_v, Wv, bv, nullptr, 1.0f, vb, L_SEQ, DMODEL, DMODEL);

  attn_kernel<<<dim3(L_SEQ / 128, NHEADS), blk, 0, stream>>>(qb, kb, vb, ab);

  oproj_kernel<<<gproj, blk, 0, stream>>>(ab, Wo, bo, (float*)d_out, L_SEQ, DMODEL, DMODEL);
}
